// MixHop_57097295233451
// MI455X (gfx1250) — matrix operand order repaired and checked by reading, not compiled, not run on hardware
//
#include <hip/hip_runtime.h>
#include <hip/hip_bf16.h>

// ---------------------------------------------------------------------------
// MixHop on gfx1250: bf16 WMMA GEMMs (2-wide N-blocked) + fp32 atomic SpMM.
// ---------------------------------------------------------------------------

typedef __attribute__((ext_vector_type(16))) __bf16 v16bf;
typedef __attribute__((ext_vector_type(8)))  __bf16 v8bf;
typedef __attribute__((ext_vector_type(8)))  float  v8f;

union V16 { v16bf v; v8bf h[2]; };

#define NN   50000
#define EE   800000
#define FIN  512
#define HHID 128
#define OC   384      // 3 * 128 concatenated width
#define CCLS 16
#define MT   3125     // 50000 / 16 exact
#define KPAD 520      // 512+8: col stride 260 dwords -> bank-conflict-free
#define KPADF 392     // 384+8 for the final (K=384) weight tile

// ---------------- elementwise converts ----------------

__global__ void cvt_f32_bf16(const float* __restrict__ in, __bf16* __restrict__ out, long n) {
    long i = (long)blockIdx.x * blockDim.x + threadIdx.x;
    if (i < n) out[i] = (__bf16)in[i];
}

__global__ void zero_f32(float* __restrict__ p, long n) {
    long i = (long)blockIdx.x * blockDim.x + threadIdx.x;
    if (i < n) p[i] = 0.f;
}

// Assemble concatenated [N,384] activation (slice0 strided + two dense hop
// slices), optional flat bias, convert to bf16 for the next WMMA GEMM.
__global__ void concat_cvt_bf16(const float* __restrict__ s0, int ld0,
                                const float* __restrict__ s1,
                                const float* __restrict__ s2,
                                const float* __restrict__ bias,  // flat [384] or null
                                __bf16* __restrict__ out, int nrows) {
    long i = (long)blockIdx.x * blockDim.x + threadIdx.x;
    long total = (long)nrows * OC;
    if (i >= total) return;
    int n = (int)(i / OC);
    int c = (int)(i - (long)n * OC);
    float v;
    if (c < 128)      v = s0[(long)n * ld0 + c];
    else if (c < 256) v = s1[(long)n * 128 + (c - 128)];
    else              v = s2[(long)n * 128 + (c - 256)];
    if (bias) v += bias[c];
    out[i] = (__bf16)v;
}

// ---------------- SpMM: out[r, yoff:yoff+128] += v * in[c, xoff:xoff+128] ----

__global__ __launch_bounds__(256) void spmm_scatter(
    const int* __restrict__ rows, const int* __restrict__ cols,
    const float* __restrict__ vals,
    const float* __restrict__ X, int ldx, int xoff,
    float* __restrict__ Y, int ldy, int yoff) {
    long tid = (long)blockIdx.x * blockDim.x + threadIdx.x;
    int e = (int)(tid >> 5);
    if (e >= EE) return;
    int j = ((int)tid & 31) * 4;
    int r = rows[e];
    int c = cols[e];
    float v = vals[e];
    const float4 xv = *(const float4*)(X + (long)c * ldx + xoff + j);
    float* yp = Y + (long)r * ldy + yoff + j;
    atomicAdd(yp + 0, v * xv.x);
    atomicAdd(yp + 1, v * xv.y);
    atomicAdd(yp + 2, v * xv.z);
    atomicAdd(yp + 3, v * xv.w);
}

// ---------------- WMMA GEMM: C[M,384] = A[M,K] @ concat_i W[i][K,128] --------
// W element (k, n): slice i = n>>7, h = n&127 -> W[i*K*128 + k*128 + h].
// grid.y indexes 32-column stripes (within one 128-col slice: 128/32 = 4).
// One block = 8 waves; wave w owns the 16x32 C tile (blockIdx.x*8+w, stripe),
// i.e. two 16x16 WMMA accumulators sharing one A fragment per k-step.
// The 32 x K weight stripe is staged transposed in LDS: wt[c][k], k contiguous.

template <int K, int MODE>  // MODE: 0 = none, 1 = bias + relu
__global__ __launch_bounds__(256) void gemm_wmma_bf16(
    const __bf16* __restrict__ A, int lda,
    const __bf16* __restrict__ W,
    const float* __restrict__ bias,   // flat [384] (MODE==1)
    float* __restrict__ C, int ldc, int mtiles) {
    __shared__ __bf16 wt[32 * KPAD];  // 33,280 B

    const int ny    = blockIdx.y;           // 32-col stripe index (0..11)
    const int slice = ny >> 2;              // which 128-col weight slice
    const int h0    = (ny & 3) << 5;        // col offset within slice
    const __bf16* wsrc = W + (long)slice * K * 128 + h0;  // (k,c) -> wsrc[k*128+c]

    {   // cooperative transpose load: vector b128 reads (8 cols), b16 scatter
        const int t  = threadIdx.x;
        const int c8 = (t & 3) * 8;         // col group 0/8/16/24
        const int kt = t >> 2;              // 64 k-threads per col group
        for (int i = 0; i < (K >> 6); ++i) {
            const int k = kt + (i << 6);
            v8bf w = *(const v8bf*)(wsrc + (long)k * 128 + c8);
#pragma unroll
            for (int j = 0; j < 8; ++j)
                wt[(c8 + j) * KPAD + k] = w[j];
        }
    }
    __syncthreads();

    const int wave = threadIdx.x >> 5;
    const int lane = threadIdx.x & 31;
    const int mtile = blockIdx.x * 8 + wave;
    if (mtile >= mtiles) return;            // wave-uniform: EXEC stays all-ones

    // A layout (matches B): lanes<16 hold K 0-15; lanes>=16 hold K 16-31
    const int kbA  = (lane < 16) ? 0 : 16;
    // B layout (matches A): lanes<16 hold K 0-15; lanes>=16 hold K 16-31
    const int kbB  = (lane < 16) ? 0 : 16;
    const int colL = lane & 15;

    const __bf16* arow  = A + (long)(mtile * 16 + (lane & 15)) * lda;
    const __bf16* bcol0 = &wt[colL * KPAD];
    const __bf16* bcol1 = &wt[(16 + colL) * KPAD];

    v8f acc0 = {};
    v8f acc1 = {};
#pragma unroll
    for (int k0 = 0; k0 < K; k0 += 32) {
        V16 a, b0, b1;
        a.h[0]  = *(const v8bf*)(arow + k0 + kbA);
        a.h[1]  = *(const v8bf*)(arow + k0 + kbA + 8);
        b0.h[0] = *(const v8bf*)(bcol0 + k0 + kbB);
        b0.h[1] = *(const v8bf*)(bcol0 + k0 + kbB + 8);
        b1.h[0] = *(const v8bf*)(bcol1 + k0 + kbB);
        b1.h[1] = *(const v8bf*)(bcol1 + k0 + kbB + 8);
        acc0 = __builtin_amdgcn_wmma_f32_16x16x32_bf16(
            false, a.v, false, b0.v, (short)0, acc0, false, false);
        acc1 = __builtin_amdgcn_wmma_f32_16x16x32_bf16(
            false, a.v, false, b1.v, (short)0, acc1, false, false);
    }

    const int r0 = mtile * 16 + ((lane < 16) ? 0 : 8);
    const int ncol0 = ny * 32 + colL;
    const int ncol1 = ncol0 + 16;
    const float bv0 = (MODE == 1) ? bias[ncol0] : 0.f;
    const float bv1 = (MODE == 1) ? bias[ncol1] : 0.f;
    float* cp0 = C + (long)r0 * ldc + ncol0;
    float* cp1 = C + (long)r0 * ldc + ncol1;
#pragma unroll
    for (int p = 0; p < 8; ++p) {           // VGPR p holds rows r0+p (ISA C/D map)
        float v0 = acc0[p] + bv0;
        float v1 = acc1[p] + bv1;
        if (MODE == 1) { v0 = fmaxf(v0, 0.f); v1 = fmaxf(v1, 0.f); }
        cp0[(long)p * ldc] = v0;
        cp1[(long)p * ldc] = v1;
    }
}

// ---------------- final: log_softmax(a2b[M,384] @ wf[384,16] + bf) -----------

__global__ __launch_bounds__(256) void final_logsoftmax_wmma(
    const __bf16* __restrict__ A,      // [M,384] bf16
    const __bf16* __restrict__ Wf,     // [384,16] bf16
    const float* __restrict__ bf_,     // [16]
    float* __restrict__ out, int mtiles) {
    __shared__ __bf16 wt[16 * KPADF];
    {
        const int t = threadIdx.x;
        const int c = t & 15;
        const int k0 = (t >> 4) * 24;      // 384/16
        for (int k = k0; k < k0 + 24; ++k)
            wt[c * KPADF + k] = Wf[k * 16 + c];
    }
    __syncthreads();

    const int wave = threadIdx.x >> 5;
    const int lane = threadIdx.x & 31;
    const int mtile = blockIdx.x * 8 + wave;
    if (mtile >= mtiles) return;

    const int kbA  = (lane < 16) ? 0 : 16;
    const int kbB  = (lane < 16) ? 0 : 16;
    const int colL = lane & 15;
    const __bf16* arow = A + (long)(mtile * 16 + (lane & 15)) * OC;
    const __bf16* bcol = &wt[colL * KPADF];

    v8f acc = {};
#pragma unroll
    for (int k0 = 0; k0 < OC; k0 += 32) {
        V16 a, b;
        a.h[0] = *(const v8bf*)(arow + k0 + kbA);
        a.h[1] = *(const v8bf*)(arow + k0 + kbA + 8);
        b.h[0] = *(const v8bf*)(bcol + k0 + kbB);
        b.h[1] = *(const v8bf*)(bcol + k0 + kbB + 8);
        acc = __builtin_amdgcn_wmma_f32_16x16x32_bf16(
            false, a.v, false, b.v, (short)0, acc, false, false);
    }

    const float bv = bf_[colL];
    const int rbase = mtile * 16 + ((lane < 16) ? 0 : 8);
#pragma unroll
    for (int p = 0; p < 8; ++p) {
        float l = acc[p] + bv;
        // row-wise reduce across the 16 lanes of this half-wave (columns)
        float m = l;
        for (int d = 1; d < 16; d <<= 1) m = fmaxf(m, __shfl_xor(m, d, 16));
        float e = __expf(l - m);
        float s = e;
        for (int d = 1; d < 16; d <<= 1) s += __shfl_xor(s, d, 16);
        out[(long)(rbase + p) * CCLS + colL] = l - m - __logf(s);
    }
}

// ---------------------------------------------------------------------------

extern "C" void kernel_launch(void* const* d_in, const int* in_sizes, int n_in,
                              void* d_out, int out_size, void* d_ws, size_t ws_size,
                              hipStream_t stream) {
    (void)in_sizes; (void)n_in; (void)out_size; (void)ws_size;
    const float* x    = (const float*)d_in[0];
    const int*   rows = (const int*)d_in[1];
    const int*   cols = rows + EE;
    const float* av   = (const float*)d_in[2];
    const float* w1   = (const float*)d_in[3];   // [3,512,128]
    const float* b1   = (const float*)d_in[4];   // flat 384
    const float* w2   = (const float*)d_in[5];   // [3,384,128]
    const float* b2   = (const float*)d_in[6];   // flat 384
    const float* wf   = (const float*)d_in[7];   // [384,16]
    const float* bfv  = (const float*)d_in[8];   // [16]

    char* ws = (char*)d_ws;
    auto carve = [&](size_t bytes) -> char* {
        char* p = ws;
        ws += (bytes + 255) & ~(size_t)255;
        return p;
    };
    __bf16* xb  = (__bf16*)carve((size_t)NN * FIN * 2);
    __bf16* w1b = (__bf16*)carve((size_t)3 * FIN * HHID * 2);
    __bf16* w2b = (__bf16*)carve((size_t)3 * OC * HHID * 2);
    __bf16* wfb = (__bf16*)carve((size_t)OC * CCLS * 2);
    float*  h1  = (float*)carve((size_t)NN * OC * 4);     // GEMM1 out, reused as g2
    float*  s1  = (float*)carve((size_t)NN * HHID * 4);
    float*  s2  = (float*)carve((size_t)NN * HHID * 4);
    float*  tmp = (float*)carve((size_t)NN * HHID * 4);
    __bf16* ab  = (__bf16*)carve((size_t)NN * OC * 2);    // a1b, reused as a2b

    const int BLK = 256;
    const int gBlocksM = (MT + 7) / 8;                       // 391
    auto g1d = [&](long n) { return (unsigned)((n + BLK - 1) / BLK); };
    const unsigned spmmGrid = g1d((long)EE * 32);
    const unsigned zeroGrid = g1d((long)NN * HHID);

    // 0) convert inputs to bf16
    cvt_f32_bf16<<<g1d((long)NN * FIN), BLK, 0, stream>>>(x, xb, (long)NN * FIN);
    cvt_f32_bf16<<<g1d((long)3 * FIN * HHID), BLK, 0, stream>>>(w1, w1b, (long)3 * FIN * HHID);
    cvt_f32_bf16<<<g1d((long)3 * OC * HHID), BLK, 0, stream>>>(w2, w2b, (long)3 * OC * HHID);
    cvt_f32_bf16<<<g1d((long)OC * CCLS), BLK, 0, stream>>>(wf, wfb, (long)OC * CCLS);

    // 1) h1 = relu(x @ concat(w1) + b1)   [N,384]
    gemm_wmma_bf16<FIN, 1><<<dim3(gBlocksM, OC / 32), BLK, 0, stream>>>(
        xb, FIN, w1b, b1, h1, OC, MT);

    // 2) hops for slices 1 (1 hop) and 2 (2 hops)
    zero_f32<<<zeroGrid, BLK, 0, stream>>>(s1, (long)NN * HHID);
    spmm_scatter<<<spmmGrid, BLK, 0, stream>>>(rows, cols, av, h1, OC, 128, s1, HHID, 0);
    zero_f32<<<zeroGrid, BLK, 0, stream>>>(tmp, (long)NN * HHID);
    spmm_scatter<<<spmmGrid, BLK, 0, stream>>>(rows, cols, av, h1, OC, 256, tmp, HHID, 0);
    zero_f32<<<zeroGrid, BLK, 0, stream>>>(s2, (long)NN * HHID);
    spmm_scatter<<<spmmGrid, BLK, 0, stream>>>(rows, cols, av, tmp, HHID, 0, s2, HHID, 0);

    // 3) a1b = bf16(concat(h1[:, :128], s1, s2))
    concat_cvt_bf16<<<g1d((long)NN * OC), BLK, 0, stream>>>(h1, OC, s1, s2, nullptr, ab, NN);

    // 4) g2 = a1b @ concat(w2)   (raw, biases added after hops) -> reuse h1
    float* g2 = h1;
    gemm_wmma_bf16<OC, 0><<<dim3(gBlocksM, OC / 32), BLK, 0, stream>>>(
        ab, OC, w2b, nullptr, g2, OC, MT);

    // 5) hops for stage-2 slices
    zero_f32<<<zeroGrid, BLK, 0, stream>>>(s1, (long)NN * HHID);
    spmm_scatter<<<spmmGrid, BLK, 0, stream>>>(rows, cols, av, g2, OC, 128, s1, HHID, 0);
    zero_f32<<<zeroGrid, BLK, 0, stream>>>(tmp, (long)NN * HHID);
    spmm_scatter<<<spmmGrid, BLK, 0, stream>>>(rows, cols, av, g2, OC, 256, tmp, HHID, 0);
    zero_f32<<<zeroGrid, BLK, 0, stream>>>(s2, (long)NN * HHID);
    spmm_scatter<<<spmmGrid, BLK, 0, stream>>>(rows, cols, av, tmp, HHID, 0, s2, HHID, 0);

    // 6) a2b = bf16(concat(g2[:, :128], s1, s2) + b2)   (reuse ab)
    concat_cvt_bf16<<<g1d((long)NN * OC), BLK, 0, stream>>>(g2, OC, s1, s2, b2, ab, NN);

    // 7) out = log_softmax(a2b @ wf + bf)
    final_logsoftmax_wmma<<<gBlocksM, BLK, 0, stream>>>(ab, wfb, bfv, (float*)d_out, MT);
}
